// GN_49606872269483
// MI455X (gfx1250) — matrix operand order repaired and checked by reading, not compiled, not run on hardware
//
#include <hip/hip_runtime.h>
#include <hip/hip_bf16.h>
#include <stdint.h>

// ---------------------------------------------------------------------------
// MeshGraphNets-style GNN on MI455X (gfx1250, wave32, WMMA bf16->f32)
// ---------------------------------------------------------------------------

typedef __bf16 bf16;
typedef bf16  v16bf __attribute__((ext_vector_type(16)));
typedef float v8f   __attribute__((ext_vector_type(8)));

#define NN 8192
#define EE 65536
#define HH 128
#define LSTEPS 15

// ------------------------- WMMA fragment loaders ---------------------------
// A operand, 16x32 bf16 tile, row-major with element stride `stride`.
// Per ISA 7.12.2: lanes 0-15 row m=lane, VGPR0-3 = K0..7, VGPR4-7 = K16..23;
// lanes 16-31 same row, K8..15 and K24..31.
__device__ __forceinline__ v16bf fragA(const bf16* tile, int stride) {
  int lane = threadIdx.x & 31;
  int m  = lane & 15;
  int kh = (lane & 16) ? 8 : 0;
  const bf16* p = tile + m * stride + kh;
  union { uint4 u[2]; v16bf v; } z;
  z.u[0] = *(const uint4*)(p);        // K kh..kh+7
  z.u[1] = *(const uint4*)(p + 16);   // K kh+16..kh+23
  return z.v;
}

// B operand, 32x16 bf16 tile given as W^T (N rows x K cols, row-major).
// Lane n%16 holds column n; lanes 0-15 carry K0..7,K16..23, lanes 16-31 K8..15,K24..31.
__device__ __forceinline__ v16bf fragB(const bf16* tileT, int stride) {
  int lane = threadIdx.x & 31;
  int n  = lane & 15;
  int kb = (lane & 16) ? 8 : 0;
  const bf16* p = tileT + n * stride + kb;
  union { uint4 u[2]; v16bf v; } z;
  z.u[0] = *(const uint4*)(p);        // K kb..kb+7
  z.u[1] = *(const uint4*)(p + 16);   // K kb+16..kb+23
  return z.v;
}

__device__ __forceinline__ uint32_t pack2(float a, float b) {
  union { bf16 h[2]; uint32_t u; } z;
  z.h[0] = (bf16)a; z.h[1] = (bf16)b;
  return z.u;
}

__device__ __forceinline__ void zero_acc(v8f acc[8]) {
#pragma unroll
  for (int i = 0; i < 8; ++i)
#pragma unroll
    for (int j = 0; j < 8; ++j) acc[i][j] = 0.f;
}

// 16 rows x 128 cols GEMM: A (16 x KC*32) bf16 in LDS, W^T (128 x KC*32) bf16
// in global. Accumulates f32 into acc[8] (8 N-tiles of 16).
template <int KC>
__device__ __forceinline__ void gemm(const bf16* __restrict__ A, int As,
                                     const bf16* __restrict__ WT, int Ws,
                                     v8f acc[8]) {
#pragma unroll
  for (int kc = 0; kc < KC; ++kc) {
    v16bf a = fragA(A + kc * 32, As);
#pragma unroll
    for (int nt = 0; nt < 8; ++nt) {
      v16bf b = fragB(WT + nt * 16 * Ws + kc * 32, Ws);
      acc[nt] = __builtin_amdgcn_wmma_f32_16x16x32_bf16(
          false, a, false, b, (short)0, acc[nt], false, false);
    }
  }
}

__device__ __forceinline__ float half16_sum(float x) {
#pragma unroll
  for (int off = 1; off < 16; off <<= 1) x += __shfl_xor(x, off, 32);
  return x;
}

// relu(acc + b1) -> 16x128 bf16 LDS tile (row-major), per C/D lane layout.
__device__ __forceinline__ void relu_bias_to_lds(const v8f acc[8],
                                                 const float* __restrict__ b1,
                                                 bf16* __restrict__ Hl) {
  int lane  = threadIdx.x & 31;
  int ncol  = lane & 15;
  int mbase = (lane & 16) ? 8 : 0;
#pragma unroll
  for (int nt = 0; nt < 8; ++nt) {
    float bb = b1[nt * 16 + ncol];
#pragma unroll
    for (int v = 0; v < 8; ++v) {
      float x = acc[nt][v] + bb;
      Hl[(mbase + v) * HH + nt * 16 + ncol] = (bf16)fmaxf(x, 0.f);
    }
  }
}

// LayerNorm across the 128 columns of each of the 16 rows, in registers.
__device__ __forceinline__ void layernorm(v8f acc[8],
                                          const float* __restrict__ g,
                                          const float* __restrict__ b) {
  int ncol = threadIdx.x & 15;
  float gv[8], bv[8];
#pragma unroll
  for (int nt = 0; nt < 8; ++nt) {
    gv[nt] = g[nt * 16 + ncol];
    bv[nt] = b[nt * 16 + ncol];
  }
#pragma unroll
  for (int v = 0; v < 8; ++v) {
    float s = 0.f, ss = 0.f;
#pragma unroll
    for (int nt = 0; nt < 8; ++nt) {
      float x = acc[nt][v];
      s += x; ss += x * x;
    }
    s  = half16_sum(s);
    ss = half16_sum(ss);
    float mean = s * (1.0f / 128.0f);
    float var  = ss * (1.0f / 128.0f) - mean * mean;
    float rstd = rsqrtf(var + 1e-5f);
#pragma unroll
    for (int nt = 0; nt < 8; ++nt)
      acc[nt][v] = (acc[nt][v] - mean) * rstd * gv[nt] + bv[nt];
  }
}

__device__ __forceinline__ void store_rows(const v8f acc[8],
                                           float* __restrict__ out,
                                           int rowBase, int rowStride) {
  int lane  = threadIdx.x & 31;
  int ncol  = lane & 15;
  int mbase = (lane & 16) ? 8 : 0;
#pragma unroll
  for (int nt = 0; nt < 8; ++nt)
#pragma unroll
    for (int v = 0; v < 8; ++v)
      out[(rowBase + mbase + v) * rowStride + nt * 16 + ncol] = acc[nt][v];
}

// ------------------------------ utility kernels -----------------------------
__global__ void k_zero(float* __restrict__ p, int n) {
  int t = blockIdx.x * blockDim.x + threadIdx.x;
  if (t < n) p[t] = 0.f;
}

// W (K x Hn, row-major f32) -> WT (Hp x Kp, row-major bf16), zero padded.
__global__ void k_convert_wT(const float* __restrict__ W, bf16* __restrict__ WT,
                             int K, int Kp, int Hn, int Hp) {
  int t = blockIdx.x * blockDim.x + threadIdx.x;
  if (t >= Hp * Kp) return;
  int n = t / Kp, k = t % Kp;
  float x = (n < Hn && k < K) ? W[k * Hn + n] : 0.f;
  WT[n * Kp + k] = (bf16)x;
}

// ------------------------------- encoders ----------------------------------
__global__ __launch_bounds__(256) void k_encode_vertex(
    const float* __restrict__ vf, const int* __restrict__ stat,
    const bf16* __restrict__ w1T, const float* __restrict__ b1,
    const bf16* __restrict__ w2T, const float* __restrict__ b2,
    const float* __restrict__ lng, const float* __restrict__ lnb,
    float* __restrict__ vout) {
  __shared__ __align__(16) bf16 A[8][16 * 32];
  __shared__ __align__(16) bf16 Hl[8][16 * HH];
  int w = threadIdx.x >> 5, lane = threadIdx.x & 31;
  int base = (blockIdx.x * 8 + w) * 16;
  bf16* Aw = A[w]; bf16* Hw = Hl[w];

  for (int t = lane; t < 16 * 32; t += 32) {   // uniform trip count
    int m = t >> 5, c = t & 31;
    int node = base + m;
    float x = 0.f;
    if (c < 2)      x = (stat[node] == c) ? 1.f : 0.f;
    else if (c < 8) x = vf[node * 6 + (c - 2)];
    Aw[t] = (bf16)x;
  }
  v8f acc[8];
  zero_acc(acc);
  gemm<1>(Aw, 32, w1T, 32, acc);
  relu_bias_to_lds(acc, b1, Hw);
  zero_acc(acc);
  gemm<4>(Hw, HH, w2T, HH, acc);
  {
    int ncol = lane & 15;
#pragma unroll
    for (int nt = 0; nt < 8; ++nt) {
      float bb = b2[nt * 16 + ncol];
#pragma unroll
      for (int v = 0; v < 8; ++v) acc[nt][v] += bb;
    }
  }
  layernorm(acc, lng, lnb);
  store_rows(acc, vout, base, HH);
}

__global__ __launch_bounds__(256) void k_encode_edge(
    const float* __restrict__ wc, const float* __restrict__ mc,
    const int* __restrict__ row, const int* __restrict__ col,
    const bf16* __restrict__ w1T, const float* __restrict__ b1,
    const bf16* __restrict__ w2T, const float* __restrict__ b2,
    const float* __restrict__ lng, const float* __restrict__ lnb,
    float* __restrict__ eout) {
  __shared__ __align__(16) bf16 A[8][16 * 32];
  __shared__ __align__(16) bf16 Hl[8][16 * HH];
  int w = threadIdx.x >> 5, lane = threadIdx.x & 31;
  int base = (blockIdx.x * 8 + w) * 16;
  bf16* Aw = A[w]; bf16* Hw = Hl[w];

  if (lane < 16) {
    int eid = base + lane;
    int r = row[eid], c = col[eid];
    float ev0 = wc[c * 2]     - wc[r * 2];
    float ev1 = wc[c * 2 + 1] - wc[r * 2 + 1];
    float mv0 = mc[c * 2]     - mc[r * 2];
    float mv1 = mc[c * 2 + 1] - mc[r * 2 + 1];
    float en = sqrtf(ev0 * ev0 + ev1 * ev1);
    float mn = sqrtf(mv0 * mv0 + mv1 * mv1);
    bf16* ar = Aw + lane * 32;
    ar[0] = (bf16)ev0; ar[1] = (bf16)ev1; ar[2] = (bf16)en;
    ar[3] = (bf16)mv0; ar[4] = (bf16)mv1; ar[5] = (bf16)mn;
#pragma unroll
    for (int c2 = 6; c2 < 32; ++c2) ar[c2] = (bf16)0.f;
  }
  v8f acc[8];
  zero_acc(acc);
  gemm<1>(Aw, 32, w1T, 32, acc);
  relu_bias_to_lds(acc, b1, Hw);
  zero_acc(acc);
  gemm<4>(Hw, HH, w2T, HH, acc);
  {
    int ncol = lane & 15;
#pragma unroll
    for (int nt = 0; nt < 8; ++nt) {
      float bb = b2[nt * 16 + ncol];
#pragma unroll
      for (int v = 0; v < 8; ++v) acc[nt][v] += bb;
    }
  }
  layernorm(acc, lng, lnb);
  store_rows(acc, eout, base, HH);
}

// ----------------------------- message passing -----------------------------
__global__ __launch_bounds__(256) void k_edge_update(
    const float* __restrict__ vbuf, float* __restrict__ ebuf,
    float* __restrict__ agg,
    const int* __restrict__ row, const int* __restrict__ col,
    const bf16* __restrict__ w1T, const float* __restrict__ b1,
    const bf16* __restrict__ w2T, const float* __restrict__ b2,
    const float* __restrict__ lng, const float* __restrict__ lnb) {
  __shared__ __align__(16) bf16 A[8][16 * 384];   // concat[v[r], v[c], e]
  __shared__ __align__(16) bf16 Hl[8][16 * HH];
  int w = threadIdx.x >> 5, lane = threadIdx.x & 31;
  int base = (blockIdx.x * 8 + w) * 16;
  bf16* Aw = A[w]; bf16* Hw = Hl[w];

  // Stage A: 16 rows x 384 cols bf16 (f32 float4 loads -> packed LDS stores)
  for (int t = lane; t < 16 * 32; t += 32) {
    int m = t >> 5, q = t & 31;
    int r = row[base + m];
    int c = col[base + m];
    float4 x = ((const float4*)(vbuf + r * HH))[q];
    uint2 p; p.x = pack2(x.x, x.y); p.y = pack2(x.z, x.w);
    *((uint2*)(Aw + m * 384 + q * 4)) = p;
    x = ((const float4*)(vbuf + c * HH))[q];
    p.x = pack2(x.x, x.y); p.y = pack2(x.z, x.w);
    *((uint2*)(Aw + m * 384 + 128 + q * 4)) = p;
    x = ((const float4*)(ebuf + (base + m) * HH))[q];
    p.x = pack2(x.x, x.y); p.y = pack2(x.z, x.w);
    *((uint2*)(Aw + m * 384 + 256 + q * 4)) = p;
  }

  v8f acc[8];
  zero_acc(acc);
  gemm<12>(Aw, 384, w1T, 384, acc);    // 384 -> 128
  relu_bias_to_lds(acc, b1, Hw);
  zero_acc(acc);
  gemm<4>(Hw, HH, w2T, HH, acc);       // 128 -> 128

  int ncol = lane & 15, mbase = (lane & 16) ? 8 : 0;
#pragma unroll
  for (int nt = 0; nt < 8; ++nt) {
    float bb = b2[nt * 16 + ncol];
#pragma unroll
    for (int v = 0; v < 8; ++v)
      acc[nt][v] += bb + ebuf[(base + mbase + v) * HH + nt * 16 + ncol];
  }
  layernorm(acc, lng, lnb);

  // write e_new and accumulate segment_sum(e_new, row) via f32 atomics (L2)
#pragma unroll
  for (int v = 0; v < 8; ++v) {
    int m = mbase + v;
    int r = row[base + m];
#pragma unroll
    for (int nt = 0; nt < 8; ++nt) {
      float x = acc[nt][v];
      ebuf[(base + m) * HH + nt * 16 + ncol] = x;
      atomicAdd(&agg[r * HH + nt * 16 + ncol], x);
    }
  }
}

__global__ __launch_bounds__(256) void k_node_update(
    float* __restrict__ vbuf, const float* __restrict__ agg,
    const bf16* __restrict__ w1T, const float* __restrict__ b1,
    const bf16* __restrict__ w2T, const float* __restrict__ b2,
    const float* __restrict__ lng, const float* __restrict__ lnb) {
  __shared__ __align__(16) bf16 A[8][16 * 256];   // concat[v, agg]
  __shared__ __align__(16) bf16 Hl[8][16 * HH];
  int w = threadIdx.x >> 5, lane = threadIdx.x & 31;
  int base = (blockIdx.x * 8 + w) * 16;
  bf16* Aw = A[w]; bf16* Hw = Hl[w];

  for (int t = lane; t < 16 * 32; t += 32) {
    int m = t >> 5, q = t & 31;
    float4 x = ((const float4*)(vbuf + (base + m) * HH))[q];
    uint2 p; p.x = pack2(x.x, x.y); p.y = pack2(x.z, x.w);
    *((uint2*)(Aw + m * 256 + q * 4)) = p;
    x = ((const float4*)(agg + (base + m) * HH))[q];
    p.x = pack2(x.x, x.y); p.y = pack2(x.z, x.w);
    *((uint2*)(Aw + m * 256 + 128 + q * 4)) = p;
  }

  v8f acc[8];
  zero_acc(acc);
  gemm<8>(Aw, 256, w1T, 256, acc);     // 256 -> 128
  relu_bias_to_lds(acc, b1, Hw);
  zero_acc(acc);
  gemm<4>(Hw, HH, w2T, HH, acc);       // 128 -> 128

  int ncol = lane & 15, mbase = (lane & 16) ? 8 : 0;
#pragma unroll
  for (int nt = 0; nt < 8; ++nt) {
    float bb = b2[nt * 16 + ncol];
#pragma unroll
    for (int v = 0; v < 8; ++v)
      acc[nt][v] += bb + vbuf[(base + mbase + v) * HH + nt * 16 + ncol];
  }
  layernorm(acc, lng, lnb);
  store_rows(acc, vbuf, base, HH);
}

// -------------------------------- decoder ----------------------------------
__global__ __launch_bounds__(256) void k_decode(
    const float* __restrict__ vbuf,
    const bf16* __restrict__ w1T, const float* __restrict__ b1,
    const bf16* __restrict__ w2T, const float* __restrict__ b2,
    float* __restrict__ out) {
  __shared__ __align__(16) bf16 A[8][16 * HH];
  __shared__ __align__(16) bf16 Hl[8][16 * HH];
  int w = threadIdx.x >> 5, lane = threadIdx.x & 31;
  int base = (blockIdx.x * 8 + w) * 16;
  bf16* Aw = A[w]; bf16* Hw = Hl[w];

  for (int t = lane; t < 16 * 32; t += 32) {
    int m = t >> 5, q = t & 31;
    float4 x = ((const float4*)(vbuf + (base + m) * HH))[q];
    uint2 p; p.x = pack2(x.x, x.y); p.y = pack2(x.z, x.w);
    *((uint2*)(Aw + m * HH + q * 4)) = p;
  }

  v8f acc[8];
  zero_acc(acc);
  gemm<4>(Aw, HH, w1T, HH, acc);       // 128 -> 128
  relu_bias_to_lds(acc, b1, Hw);

  // Second layer: single 16-wide N tile (cols 0..5 valid, rest zero-padded W)
  v8f a1;
#pragma unroll
  for (int j = 0; j < 8; ++j) a1[j] = 0.f;
#pragma unroll
  for (int kc = 0; kc < 4; ++kc) {
    v16bf a = fragA(Hw + kc * 32, HH);
    v16bf b = fragB(w2T + kc * 32, HH);
    a1 = __builtin_amdgcn_wmma_f32_16x16x32_bf16(false, a, false, b, (short)0,
                                                 a1, false, false);
  }
  int ncol = lane & 15, mbase = (lane & 16) ? 8 : 0;
  if (ncol < 6) {
    float bb = b2[ncol];
#pragma unroll
    for (int v = 0; v < 8; ++v)
      out[(base + mbase + v) * 6 + ncol] = a1[v] + bb;
  }
}

// ------------------------------ host launcher ------------------------------
extern "C" void kernel_launch(void* const* d_in, const int* in_sizes, int n_in,
                              void* d_out, int out_size, void* d_ws,
                              size_t ws_size, hipStream_t stream) {
  const float* world = (const float*)d_in[0];
  const float* vfeat = (const float*)d_in[1];
  const float* meshc = (const float*)d_in[2];
  const float* ve_w1 = (const float*)d_in[3];  const float* ve_b1 = (const float*)d_in[4];
  const float* ve_w2 = (const float*)d_in[5];  const float* ve_b2 = (const float*)d_in[6];
  const float* ee_w1 = (const float*)d_in[7];  const float* ee_b1 = (const float*)d_in[8];
  const float* ee_w2 = (const float*)d_in[9];  const float* ee_b2 = (const float*)d_in[10];
  const float* enc_g = (const float*)d_in[11]; const float* enc_b = (const float*)d_in[12];
  const float* em_w1 = (const float*)d_in[13]; const float* em_b1 = (const float*)d_in[14];
  const float* em_w2 = (const float*)d_in[15]; const float* em_b2 = (const float*)d_in[16];
  const float* em_g  = (const float*)d_in[17]; const float* em_bb = (const float*)d_in[18];
  const float* nm_w1 = (const float*)d_in[19]; const float* nm_b1 = (const float*)d_in[20];
  const float* nm_w2 = (const float*)d_in[21]; const float* nm_b2 = (const float*)d_in[22];
  const float* nm_g  = (const float*)d_in[23]; const float* nm_bb = (const float*)d_in[24];
  const float* dc_w1 = (const float*)d_in[25]; const float* dc_b1 = (const float*)d_in[26];
  const float* dc_w2 = (const float*)d_in[27]; const float* dc_b2 = (const float*)d_in[28];
  const int* eidx = (const int*)d_in[29];
  const int* statn = (const int*)d_in[30];
  const int* row = eidx;       // edge_index[0,0,:]
  const int* col = eidx + EE;  // edge_index[0,1,:]

  char* p = (char*)d_ws;
  float* vbuf = (float*)p; p += (size_t)NN * HH * 4;
  float* agg  = (float*)p; p += (size_t)NN * HH * 4;
  float* ebuf = (float*)p; p += (size_t)EE * HH * 4;
  bf16* ve_w1T = (bf16*)p; p += 128 * 32 * 2;
  bf16* ve_w2T = (bf16*)p; p += 128 * 128 * 2;
  bf16* ee_w1T = (bf16*)p; p += 128 * 32 * 2;
  bf16* ee_w2T = (bf16*)p; p += 128 * 128 * 2;
  bf16* em_w1T = (bf16*)p; p += 128 * 384 * 2;
  bf16* em_w2T = (bf16*)p; p += 128 * 128 * 2;
  bf16* nm_w1T = (bf16*)p; p += 128 * 256 * 2;
  bf16* nm_w2T = (bf16*)p; p += 128 * 128 * 2;
  bf16* dc_w1T = (bf16*)p; p += 128 * 128 * 2;
  bf16* dc_w2T = (bf16*)p; p += 16 * 128 * 2;

  auto conv = [&](const float* W, bf16* WT, int K, int Kp, int Hn, int Hp) {
    int tot = Hp * Kp;
    k_convert_wT<<<(tot + 255) / 256, 256, 0, stream>>>(W, WT, K, Kp, Hn, Hp);
  };
  conv(ve_w1, ve_w1T, 8, 32, 128, 128);
  conv(ve_w2, ve_w2T, 128, 128, 128, 128);
  conv(ee_w1, ee_w1T, 6, 32, 128, 128);
  conv(ee_w2, ee_w2T, 128, 128, 128, 128);
  conv(em_w1, em_w1T, 384, 384, 128, 128);
  conv(em_w2, em_w2T, 128, 128, 128, 128);
  conv(nm_w1, nm_w1T, 256, 256, 128, 128);
  conv(nm_w2, nm_w2T, 128, 128, 128, 128);
  conv(dc_w1, dc_w1T, 128, 128, 128, 128);
  conv(dc_w2, dc_w2T, 128, 128, 6, 16);

  k_encode_vertex<<<NN / 128, 256, 0, stream>>>(vfeat, statn, ve_w1T, ve_b1,
                                                ve_w2T, ve_b2, enc_g, enc_b, vbuf);
  k_encode_edge<<<EE / 128, 256, 0, stream>>>(world, meshc, row, col, ee_w1T,
                                              ee_b1, ee_w2T, ee_b2, enc_g,
                                              enc_b, ebuf);

  for (int s = 0; s < LSTEPS; ++s) {
    k_zero<<<(NN * HH + 255) / 256, 256, 0, stream>>>(agg, NN * HH);
    k_edge_update<<<EE / 128, 256, 0, stream>>>(vbuf, ebuf, agg, row, col,
                                                em_w1T, em_b1, em_w2T, em_b2,
                                                em_g, em_bb);
    k_node_update<<<NN / 128, 256, 0, stream>>>(vbuf, agg, nm_w1T, nm_b1,
                                                nm_w2T, nm_b2, nm_g, nm_bb);
  }
  k_decode<<<NN / 128, 256, 0, stream>>>(vbuf, dc_w1T, dc_b1, dc_w2T, dc_b2,
                                         (float*)d_out);
}
